// DYCEP_8572754723266
// MI455X (gfx1250) — matrix operand order repaired and checked by reading, not compiled, not run on hardware
//
#include <hip/hip_runtime.h>
#include <hip/hip_bf16.h>

// ---------------------------------------------------------------------------
// CNN(3 conv, stride2) -> fc(32->256) -> 6x Mamba -> rmsnorm -> head -> softmax
// GEMMs: v_wmma_f32_16x16x32_bf16 (f32 accumulate), 128x64 block tile,
//        double-buffered LDS (1 barrier / K-step), register-prefetched global
//        loads overlapped with WMMA, branch-free clamped+masked N padding.
// Scan:  double-buffered global_load_async_to_lds_b32 pipeline (ASYNCcnt).
// ---------------------------------------------------------------------------

typedef __bf16 bf16_t;
typedef __attribute__((ext_vector_type(16))) __bf16 v16bf;
typedef __attribute__((ext_vector_type(4)))  __bf16 v4bf;
typedef __attribute__((ext_vector_type(8)))  float  v8f;

#define Bn 8
#define Tn 256
#define DM 256
#define DI 512
#define DS 16
#define DTR 16
#define BT (Bn * Tn)   // 2048 rows

__device__ __forceinline__ float silu_f(float x) { return x / (1.0f + __expf(-x)); }
__device__ __forceinline__ float gelu_f(float x) {
    float x3 = x * x * x;
    return 0.5f * x * (1.0f + tanhf(0.7978845608f * (x + 0.044715f * x3)));
}
__device__ __forceinline__ float softplus_f(float x) {
    return (x > 20.0f) ? x : log1pf(__expf(x));
}
__device__ __forceinline__ v4bf cvt_bf16x4(float4 f) {
    v4bf h;
    h[0] = (bf16_t)f.x; h[1] = (bf16_t)f.y;
    h[2] = (bf16_t)f.z; h[3] = (bf16_t)f.w;
    return h;
}

// ---------------------------------------------------------------------------
// Fused spatial encoder: one block per frame. conv1/2/3 staged in LDS (f16),
// global average pool -> z1 (BT x 32).
// ---------------------------------------------------------------------------
__global__ __launch_bounds__(256) void encoder_kernel(
    const float* __restrict__ x,
    const float* __restrict__ w1, const float* __restrict__ b1,
    const float* __restrict__ w2, const float* __restrict__ b2,
    const float* __restrict__ w3, const float* __restrict__ b3,
    float* __restrict__ z1)
{
    __shared__ _Float16 sC1[16 * 32 * 32];  // 32 KB
    __shared__ _Float16 sC2[32 * 16 * 16];  // 16 KB
    __shared__ float    sPool[32 * 64];     //  8 KB
    const int tid = threadIdx.x;
    const int frame = blockIdx.x;
    const float* in = x + (size_t)frame * 64 * 64;

    // conv1: 1->16, 64x64 -> 32x32, SAME pad (lo=0, hi=1), relu
    for (int i = tid; i < 16 * 32 * 32; i += 256) {
        int oc = i >> 10, p = i & 1023, oy = p >> 5, ox = p & 31;
        float acc = b1[oc];
        #pragma unroll
        for (int kh = 0; kh < 3; ++kh) {
            int iy = 2 * oy + kh;
            if (iy >= 64) continue;
            #pragma unroll
            for (int kw = 0; kw < 3; ++kw) {
                int ix = 2 * ox + kw;
                if (ix >= 64) continue;
                acc += w1[oc * 9 + kh * 3 + kw] * in[iy * 64 + ix];
            }
        }
        sC1[i] = (_Float16)fmaxf(acc, 0.0f);
    }
    __syncthreads();

    // conv2: 16->32, 32x32 -> 16x16
    for (int i = tid; i < 32 * 16 * 16; i += 256) {
        int oc = i >> 8, p = i & 255, oy = p >> 4, ox = p & 15;
        float acc = b2[oc];
        for (int ic = 0; ic < 16; ++ic) {
            const float* wp = w2 + (oc * 16 + ic) * 9;
            const _Float16* ip = sC1 + ic * 1024;
            #pragma unroll
            for (int kh = 0; kh < 3; ++kh) {
                int iy = 2 * oy + kh;
                if (iy >= 32) continue;
                #pragma unroll
                for (int kw = 0; kw < 3; ++kw) {
                    int ix = 2 * ox + kw;
                    if (ix >= 32) continue;
                    acc += wp[kh * 3 + kw] * (float)ip[iy * 32 + ix];
                }
            }
        }
        sC2[i] = (_Float16)fmaxf(acc, 0.0f);
    }
    __syncthreads();

    // conv3: 32->32, 16x16 -> 8x8, relu -> sPool
    for (int i = tid; i < 32 * 8 * 8; i += 256) {
        int oc = i >> 6, p = i & 63, oy = p >> 3, ox = p & 7;
        float acc = b3[oc];
        for (int ic = 0; ic < 32; ++ic) {
            const float* wp = w3 + (oc * 32 + ic) * 9;
            const _Float16* ip = sC2 + ic * 256;
            #pragma unroll
            for (int kh = 0; kh < 3; ++kh) {
                int iy = 2 * oy + kh;
                if (iy >= 16) continue;
                #pragma unroll
                for (int kw = 0; kw < 3; ++kw) {
                    int ix = 2 * ox + kw;
                    if (ix >= 16) continue;
                    acc += wp[kh * 3 + kw] * (float)ip[iy * 16 + ix];
                }
            }
        }
        sPool[i] = fmaxf(acc, 0.0f);
    }
    __syncthreads();

    if (tid < 32) {
        float s = 0.0f;
        #pragma unroll
        for (int j = 0; j < 64; ++j) s += sPool[tid * 64 + j];
        z1[frame * 32 + tid] = s * (1.0f / 64.0f);
    }
}

// ---------------------------------------------------------------------------
// WMMA GEMM: C[M,N] = act(A[M,K] @ W[N,K]^T + bias) + residual
// Block tile 128(M) x 64(N), K step 32, 8 waves; each wave: 1 A fragment x
// 4 B fragments -> 4 x v_wmma_f32_16x16x32_bf16 per K-step.
// Double-buffered LDS: tile k+1 is global-loaded into registers while the
// WMMAs consume tile k from LDS; single barrier per K-step.
// N padding is branch-free: clamped row address + multiplicative zero mask.
// Requires M % 128 == 0, K % 32 == 0 (true for all launches).
// act: 0=none, 1=gelu.
// ---------------------------------------------------------------------------
#define GBM 128
#define GBN 64
#define GBK 32

__global__ __launch_bounds__(256) void gemm_wmma_kernel(
    const float* __restrict__ A, const float* __restrict__ W,
    const float* __restrict__ bias, const float* __restrict__ residual,
    float* __restrict__ C, int M, int N, int K, int act)
{
    __shared__ bf16_t sA[2][GBM][GBK];   // 16 KB
    __shared__ bf16_t sW[2][GBN][GBK];   //  8 KB
    const int tid  = threadIdx.x;
    const int wv   = tid >> 5;           // wave 0..7
    const int lane = tid & 31;
    const int m0 = blockIdx.x * GBM;
    const int n0 = blockIdx.y * GBN;

    float4 ra[4];  // A tile: 128x32 = 1024 float4, 4 per thread
    float4 rw[2];  // W tile:  64x32 =  512 float4, 2 per thread

    v8f acc[4] = {{}, {}, {}, {}};

    auto load_tiles = [&](int k0) {
        #pragma unroll
        for (int i = 0; i < 4; ++i) {
            int e4 = tid + i * 256, r = e4 >> 3, c = (e4 & 7) << 2;
            ra[i] = *(const float4*)(A + (size_t)(m0 + r) * K + k0 + c);
        }
        #pragma unroll
        for (int i = 0; i < 2; ++i) {
            int e4 = tid + i * 256, r = e4 >> 3, c = (e4 & 7) << 2;
            int n = n0 + r;
            int nc = (n < N) ? n : (N - 1);          // clamp: always valid
            float4 f = *(const float4*)(W + (size_t)nc * K + k0 + c);
            float msk = (n < N) ? 1.0f : 0.0f;       // zero padded rows
            f.x *= msk; f.y *= msk; f.z *= msk; f.w *= msk;
            rw[i] = f;
        }
    };
    auto store_tiles = [&](int buf) {
        #pragma unroll
        for (int i = 0; i < 4; ++i) {
            int e4 = tid + i * 256, r = e4 >> 3, c = (e4 & 7) << 2;
            *(v4bf*)&sA[buf][r][c] = cvt_bf16x4(ra[i]);
        }
        #pragma unroll
        for (int i = 0; i < 2; ++i) {
            int e4 = tid + i * 256, r = e4 >> 3, c = (e4 & 7) << 2;
            *(v4bf*)&sW[buf][r][c] = cvt_bf16x4(rw[i]);
        }
    };

    load_tiles(0);
    store_tiles(0);
    __syncthreads();

    const int nk = K >> 5;
    for (int kt = 0; kt < nk; ++kt) {
        const int cur = kt & 1;
        const bool more = (kt + 1) < nk;
        if (more) load_tiles((kt + 1) << 5);   // global -> regs (overlaps WMMA)

        // A fragment (ISA 16-bit A 16x32 layout):
        // lanes 0-15 : row=lane,    K = 0..7  then 16..23
        // lanes 16-31: row=lane-16, K = 8..15 then 24..31
        const int arow = (wv << 4) + (lane & 15);
        const int kb   = (lane >> 4) << 3;
        v16bf a;
        #pragma unroll
        for (int j = 0; j < 8; ++j) {
            a[j]     = sA[cur][arow][kb + j];
            a[8 + j] = sA[cur][arow][kb + 16 + j];
        }
        // B fragments (32x16 K x N): lane&15 = column, K order matches A fragment
        const int klo = (lane >> 4) << 3;
        const int nn  = lane & 15;
        v16bf bf[4];
        #pragma unroll
        for (int t = 0; t < 4; ++t)
            #pragma unroll
            for (int j = 0; j < 16; ++j)
                bf[t][j] = sW[cur][t * 16 + nn][klo + (j & 7) + ((j >> 3) << 4)];

        #pragma unroll
        for (int t = 0; t < 4; ++t)
            acc[t] = __builtin_amdgcn_wmma_f32_16x16x32_bf16(
                false, a, false, bf[t], (short)0, acc[t], false, false);

        if (more) store_tiles(cur ^ 1);        // regs -> other LDS buffer
        __syncthreads();  // next buffer published AND current fully consumed
    }

    // D layout: VGPR j -> M = j (lanes 0-15) / j+8 (lanes 16-31); N = lane&15
    const int rbase = m0 + (wv << 4) + ((lane >> 4) << 3);
    #pragma unroll
    for (int t = 0; t < 4; ++t) {
        const int cc = n0 + t * 16 + (lane & 15);
        if (cc >= N) continue;
        #pragma unroll
        for (int j = 0; j < 8; ++j) {
            const int r = rbase + j;
            float v = acc[t][j];
            if (bias) v += bias[cc];
            if (act == 1) v = gelu_f(v);
            if (residual) v += residual[(size_t)r * N + cc];
            C[(size_t)r * N + cc] = v;
        }
    }
}

// ---------------------------------------------------------------------------
// RMSNorm over last dim (256), one block per row
// ---------------------------------------------------------------------------
__global__ __launch_bounds__(256) void rmsnorm_kernel(
    const float* __restrict__ in, const float* __restrict__ w, float* __restrict__ out)
{
    __shared__ float red[256];
    const int row = blockIdx.x, t = threadIdx.x;
    float v = in[(size_t)row * DM + t];
    red[t] = v * v;
    __syncthreads();
    for (int s = 128; s > 0; s >>= 1) {
        if (t < s) red[t] += red[t + s];
        __syncthreads();
    }
    float scale = rsqrtf(red[0] * (1.0f / DM) + 1e-5f);
    out[(size_t)row * DM + t] = v * scale * w[t];
}

// ---------------------------------------------------------------------------
// Causal depthwise conv1d (K=4) over T + SiLU. xz row stride 1024
// ---------------------------------------------------------------------------
__global__ __launch_bounds__(256) void dwconv_kernel(
    const float* __restrict__ xz, const float* __restrict__ cw,
    const float* __restrict__ cb, float* __restrict__ xact)
{
    int idx = blockIdx.x * 256 + threadIdx.x;       // < BT*DI
    int e = idx & (DI - 1);
    int bt = idx >> 9;
    int t = bt & (Tn - 1);
    float acc = cb[e];
    #pragma unroll
    for (int k = 0; k < 4; ++k) {
        int tt = t - 3 + k;
        if (tt >= 0) acc += cw[e * 4 + k] * xz[(size_t)(bt - 3 + k) * (2 * DI) + e];
    }
    xact[idx] = silu_f(acc);
}

// ---------------------------------------------------------------------------
// delta = softplus(dt @ dt_proj_w^T + dt_proj_b); dt = dbc[:, 0:16]
// ---------------------------------------------------------------------------
__global__ __launch_bounds__(256) void delta_kernel(
    const float* __restrict__ dbc, const float* __restrict__ dpw,
    const float* __restrict__ dpb, float* __restrict__ delta)
{
    int idx = blockIdx.x * 256 + threadIdx.x;       // < BT*DI
    int e = idx & (DI - 1);
    int bt = idx >> 9;
    const float* dt = dbc + (size_t)bt * 48;
    float acc = dpb[e];
    #pragma unroll
    for (int r = 0; r < DTR; ++r) acc += dt[r] * dpw[e * DTR + r];
    delta[idx] = softplus_f(acc);
}

// ---------------------------------------------------------------------------
// Selective scan + D-skip + SiLU(z) gate. One thread per (b,e) chain.
// The per-timestep B/C vector (32 floats, shared by all 512 channels) is
// staged one step ahead via global_load_async_to_lds_b32 (ASYNCcnt),
// double-buffered in LDS: wave 0 issues 32 lanes x 4B = 128B per step.
// grid (B, DI/256), block 256
// ---------------------------------------------------------------------------
__global__ __launch_bounds__(256) void scan_kernel(
    const float* __restrict__ xact, const float* __restrict__ xz,
    const float* __restrict__ dbc,  const float* __restrict__ delta,
    const float* __restrict__ a_log, const float* __restrict__ Dp,
    float* __restrict__ y)
{
    const int b = blockIdx.x;
    const int e = blockIdx.y * 256 + threadIdx.x;   // 0..511
    __shared__ float sBC[2][2 * DS];

    float Arow[DS];
    #pragma unroll
    for (int n = 0; n < DS; ++n) Arow[n] = -__expf(a_log[e * DS + n]);
    const float Dv = Dp[e];
    float s[DS];
    #pragma unroll
    for (int n = 0; n < DS; ++n) s[n] = 0.0f;

    // issue async copy of B/C vector for t = 0 (wave 0: lanes 0..31)
    if (threadIdx.x < 32) {
        unsigned ldsoff = (unsigned)(size_t)&sBC[0][threadIdx.x];
        unsigned long long g =
            (unsigned long long)(size_t)(dbc + (size_t)(b * Tn) * 48 + DTR + threadIdx.x);
        asm volatile("global_load_async_to_lds_b32 %0, %1, off"
                     :: "v"(ldsoff), "v"(g) : "memory");
    }

    for (int t = 0; t < Tn; ++t) {
        const int bt = b * Tn + t;
        asm volatile("s_wait_asynccnt 0x0" ::: "memory");
        __syncthreads();   // B/C for t landed; all waves done with t-1
        // prefetch B/C for t+1 into the other buffer
        if (t + 1 < Tn && threadIdx.x < 32) {
            unsigned ldsoff = (unsigned)(size_t)&sBC[(t + 1) & 1][threadIdx.x];
            unsigned long long g =
                (unsigned long long)(size_t)(dbc + (size_t)(bt + 1) * 48 + DTR + threadIdx.x);
            asm volatile("global_load_async_to_lds_b32 %0, %1, off"
                         :: "v"(ldsoff), "v"(g) : "memory");
        }
        const float* Bv = sBC[t & 1];
        const float* Cv = Bv + DS;
        float d = delta[(size_t)bt * DI + e];
        float u = xact[(size_t)bt * DI + e];
        float yv = 0.0f;
        #pragma unroll
        for (int n = 0; n < DS; ++n) {
            s[n] = __expf(d * Arow[n]) * s[n] + d * Bv[n] * u;
            yv += s[n] * Cv[n];
        }
        yv += u * Dv;
        float zv = xz[(size_t)bt * (2 * DI) + DI + e];
        yv *= silu_f(zv);
        y[(size_t)bt * DI + e] = yv;
    }
}

// ---------------------------------------------------------------------------
// head2: logits[row] = hh[row, :64] . w2 + b2   (grid BT, block 64)
// ---------------------------------------------------------------------------
__global__ __launch_bounds__(64) void head2_kernel(
    const float* __restrict__ hh, const float* __restrict__ w2,
    const float* __restrict__ b2, float* __restrict__ logits)
{
    __shared__ float red[64];
    const int row = blockIdx.x, t = threadIdx.x;
    red[t] = hh[(size_t)row * 64 + t] * w2[t];
    __syncthreads();
    for (int s = 32; s > 0; s >>= 1) {
        if (t < s) red[t] += red[t + s];
        __syncthreads();
    }
    if (t == 0) logits[row] = red[0] + b2[0];
}

// ---------------------------------------------------------------------------
// softmax over T per batch; zero t==0 AFTER normalization (per reference)
// ---------------------------------------------------------------------------
__global__ __launch_bounds__(256) void softmax_kernel(
    const float* __restrict__ logits, float* __restrict__ out)
{
    __shared__ float red[256];
    const int b = blockIdx.x, t = threadIdx.x;
    float v = logits[b * Tn + t];
    red[t] = v;
    __syncthreads();
    for (int s = 128; s > 0; s >>= 1) {
        if (t < s) red[t] = fmaxf(red[t], red[t + s]);
        __syncthreads();
    }
    float mx = red[0];
    __syncthreads();
    float e = __expf(v - mx);
    red[t] = e;
    __syncthreads();
    for (int s = 128; s > 0; s >>= 1) {
        if (t < s) red[t] += red[t + s];
        __syncthreads();
    }
    float sum = red[0];
    out[b * Tn + t] = (t == 0) ? 0.0f : e / sum;
}

// ---------------------------------------------------------------------------
extern "C" void kernel_launch(void* const* d_in, const int* in_sizes, int n_in,
                              void* d_out, int out_size, void* d_ws, size_t ws_size,
                              hipStream_t stream)
{
    (void)in_sizes; (void)n_in; (void)out_size; (void)ws_size;
    const float* x       = (const float*)d_in[0];
    const float* cnn_w1  = (const float*)d_in[1];
    const float* cnn_b1  = (const float*)d_in[2];
    const float* cnn_w2  = (const float*)d_in[3];
    const float* cnn_b2  = (const float*)d_in[4];
    const float* cnn_w3  = (const float*)d_in[5];
    const float* cnn_b3  = (const float*)d_in[6];
    const float* fc_w    = (const float*)d_in[7];
    const float* fc_b    = (const float*)d_in[8];
    const float* norm_w  = (const float*)d_in[9];
    const float* ipw     = (const float*)d_in[10];
    const float* cw      = (const float*)d_in[11];
    const float* cb      = (const float*)d_in[12];
    const float* xpw     = (const float*)d_in[13];
    const float* dtw     = (const float*)d_in[14];
    const float* dtb     = (const float*)d_in[15];
    const float* a_log   = (const float*)d_in[16];
    const float* Dp      = (const float*)d_in[17];
    const float* opw     = (const float*)d_in[18];
    const float* normf_w = (const float*)d_in[19];
    const float* hw1     = (const float*)d_in[20];
    const float* hb1     = (const float*)d_in[21];
    const float* hw2     = (const float*)d_in[22];
    const float* hb2     = (const float*)d_in[23];
    float* out = (float*)d_out;

    // workspace carve-out (floats)
    float* ws = (float*)d_ws;
    size_t off = 0;
    float* z1     = ws + off; off += (size_t)BT * 32;       // 2048 x 32
    float* h      = ws + off; off += (size_t)BT * DM;       // 2048 x 256
    float* normed = ws + off; off += (size_t)BT * DM;       // 2048 x 256
    float* xz     = ws + off; off += (size_t)BT * 2 * DI;   // 2048 x 1024
    float* xact   = ws + off; off += (size_t)BT * DI;       // 2048 x 512
    float* dbc    = ws + off; off += (size_t)BT * 48;       // 2048 x 48
    float* delta  = ws + off; off += (size_t)BT * DI;       // 2048 x 512
    float* ybuf   = ws + off; off += (size_t)BT * DI;       // 2048 x 512
    float* hh     = ws + off; off += (size_t)BT * 64;       // 2048 x 64
    float* logits = ws + off; off += (size_t)BT;            // 2048

    // 1) spatial encoder
    encoder_kernel<<<BT, 256, 0, stream>>>(x, cnn_w1, cnn_b1, cnn_w2, cnn_b2,
                                           cnn_w3, cnn_b3, z1);
    // 2) fc: h = z1 @ fc_w^T + fc_b        (M=2048, N=256, K=32)
    gemm_wmma_kernel<<<dim3(BT / GBM, DM / GBN), 256, 0, stream>>>(
        z1, fc_w, fc_b, nullptr, h, BT, DM, 32, 0);

    // 3) 6 Mamba layers
    for (int i = 0; i < 6; ++i) {
        rmsnorm_kernel<<<BT, 256, 0, stream>>>(h, norm_w + (size_t)i * DM, normed);
        // xz = normed @ in_proj^T          (N=1024, K=256)
        gemm_wmma_kernel<<<dim3(BT / GBM, (2 * DI) / GBN), 256, 0, stream>>>(
            normed, ipw + (size_t)i * 2 * DI * DM, nullptr, nullptr,
            xz, BT, 2 * DI, DM, 0);
        dwconv_kernel<<<(BT * DI) / 256, 256, 0, stream>>>(
            xz, cw + (size_t)i * DI * 4, cb + (size_t)i * DI, xact);
        // dbc = xact @ x_proj^T            (N=48 padded to 64, K=512)
        gemm_wmma_kernel<<<dim3(BT / GBM, 1), 256, 0, stream>>>(
            xact, xpw + (size_t)i * 48 * DI, nullptr, nullptr,
            dbc, BT, 48, DI, 0);
        delta_kernel<<<(BT * DI) / 256, 256, 0, stream>>>(
            dbc, dtw + (size_t)i * DI * DTR, dtb + (size_t)i * DI, delta);
        scan_kernel<<<dim3(Bn, DI / 256), 256, 0, stream>>>(
            xact, xz, dbc, delta, a_log + (size_t)i * DI * DS, Dp + (size_t)i * DI, ybuf);
        // h = h + ybuf @ out_proj^T        (N=256, K=512), residual fused
        gemm_wmma_kernel<<<dim3(BT / GBM, DM / GBN), 256, 0, stream>>>(
            ybuf, opw + (size_t)i * DM * DI, nullptr, h, h, BT, DM, DI, 0);
    }

    // 4) final norm + head
    rmsnorm_kernel<<<BT, 256, 0, stream>>>(h, normf_w, normed);
    gemm_wmma_kernel<<<dim3(BT / GBM, 1), 256, 0, stream>>>(
        normed, hw1, hb1, nullptr, hh, BT, 64, DM, 1);
    head2_kernel<<<BT, 64, 0, stream>>>(hh, hw2, hb2, logits);
    softmax_kernel<<<Bn, 256, 0, stream>>>(logits, out);
}
